// SpellerModel_3779571220493
// MI455X (gfx1250) — matrix operand order repaired and checked by reading, not compiled, not run on hardware
//
#include <hip/hip_runtime.h>
#include <hip/hip_bf16.h>

// ---------------------------------------------------------------------------
// Speller (3-layer LSTM + attention decoder) for MI455X / gfx1250.
// Strategy: one-time prepass converts all weights and attention K/V to bf16
// (total ~100MB -> resident in 192MB L2). Per timestep: 3 fused
// WMMA-bf16 LSTM kernels (GEMM + cell update in epilogue), 1 WMMA proj1
// GEMM, 1 per-batch attention/softmax/context/vocab kernel. h state double
// buffered (bf16), c state fp32 in place.
// ---------------------------------------------------------------------------

typedef __attribute__((ext_vector_type(16))) __bf16 v16bf;
typedef __attribute__((ext_vector_type(8)))  __bf16 bf16x8;
typedef __attribute__((ext_vector_type(8)))  float  v8f;

#define T_STEPS 300
#define BATCH   64
#define S_ENC   400
#define HID     1024
#define CTX     512
#define EMB     256
#define VOCAB   34
#define KIN1    (EMB + CTX)   // 768
#define G4      (4 * HID)     // 4096
#define KV_ELEMS (BATCH * S_ENC * CTX)  // 13,107,200

// ---------------------------------------------------------------------------
// WMMA helpers (CDNA5 16x16x32 bf16 -> f32)
// ---------------------------------------------------------------------------
static __device__ __forceinline__ v8f wmma_bf16f32(v16bf a, v16bf b, v8f c) {
  // (neg_a, A, neg_b, B, c_mod, C, reuse_a, reuse_b)
  return __builtin_amdgcn_wmma_f32_16x16x32_bf16(false, a, false, b,
                                                 (short)0, c, false, false);
}

// A-matrix 16x32 bf16 fragment.  ISA 7.12.2: lanes 0-15 hold rows M=0..15
// with K k0+0..7 and k0+16..23; lanes 16-31 hold the same rows with
// K k0+8..15 and k0+24..31.  Two contiguous 16B loads per lane.
static __device__ __forceinline__
v16bf load_a_frag(const __bf16* __restrict__ X, int ld, int mbase, int k0, int lane) {
  const int m  = mbase + (lane & 15);
  const int kk = k0 + ((lane & 16) ? 8 : 0);
  const __bf16* p = X + (size_t)m * ld + kk;
  bf16x8 lo = *reinterpret_cast<const bf16x8*>(p);
  bf16x8 hi = *reinterpret_cast<const bf16x8*>(p + 16);
  return __builtin_shufflevector(lo, hi, 0,1,2,3,4,5,6,7,8,9,10,11,12,13,14,15);
}

// B-matrix 32x16 bf16 fragment for W^T GEMM: B[k][n] = W[colbase+n][k].
// Lanes 0-15 hold column n with K k0+0..7 and k0+16..23; lanes 16-31 hold
// K k0+8..15 and k0+24..31.  Two 16B loads per lane from W row (colbase+n).
static __device__ __forceinline__
v16bf load_b_frag(const __bf16* __restrict__ W, int ld, int colbase, int k0, int lane) {
  const int n  = lane & 15;
  const int kk = k0 + ((lane & 16) ? 8 : 0);
  const __bf16* p = W + (size_t)(colbase + n) * ld + kk;
  bf16x8 lo = *reinterpret_cast<const bf16x8*>(p);
  bf16x8 hi = *reinterpret_cast<const bf16x8*>(p + 16);
  return __builtin_shufflevector(lo, hi, 0,1,2,3,4,5,6,7,8,9,10,11,12,13,14,15);
}

static __device__ __forceinline__ float sigmoidf_(float x) {
  return 1.0f / (1.0f + __expf(-x));
}

// ---------------------------------------------------------------------------
// Fused LSTM cell:  gates = X @ Wih^T + Hprev @ Whh^T + b;  cell update in
// epilogue.  Block = 256 thr (8 waves).  Block owns 32 hidden units; each
// wave owns (mtile, 16-unit group) and accumulates the matching 16x16 tile
// in ALL FOUR gate blocks so i/f/g/o stay wave-local for the cell update.
// Grid: HID/32 = 32 blocks.
// ---------------------------------------------------------------------------
__global__ void __launch_bounds__(256)
k_lstm(const __bf16* __restrict__ X, int ldx,
       const __bf16* __restrict__ Wih,          // (4*HID) x ldx, bf16
       const __bf16* __restrict__ Hprev,        // BATCH x HID, bf16
       const __bf16* __restrict__ Whh,          // (4*HID) x HID, bf16
       const float*  __restrict__ b_ih, const float* __restrict__ b_hh,
       float*        __restrict__ Cst,          // BATCH x HID, f32 (in place)
       __bf16*       __restrict__ Hout)         // BATCH x HID, bf16
{
  const int lane  = threadIdx.x & 31;
  const int w     = threadIdx.x >> 5;   // wave 0..7 (wave32)
  const int mtile = w & 3;              // batch-row tile (4 x 16 = 64)
  const int ug    = w >> 2;             // unit group 0..1
  const int ub    = blockIdx.x * 32 + ug * 16;   // 16 hidden units
  const int mbase = mtile * 16;

  v8f acc[4];
  #pragma unroll
  for (int g = 0; g < 4; ++g) acc[g] = v8f{0.f,0.f,0.f,0.f,0.f,0.f,0.f,0.f};

  // ---- X @ Wih^T ----
  for (int k0 = 0; k0 < ldx; k0 += 32) {
    v16bf a = load_a_frag(X, ldx, mbase, k0, lane);
    __builtin_prefetch(X + (size_t)(mbase + (lane & 15)) * ldx + k0 + 32, 0, 3);
    #pragma unroll
    for (int g = 0; g < 4; ++g) {
      v16bf b = load_b_frag(Wih, ldx, g * HID + ub, k0, lane);
      acc[g] = wmma_bf16f32(a, b, acc[g]);
    }
  }
  // ---- Hprev @ Whh^T ----
  for (int k0 = 0; k0 < HID; k0 += 32) {
    v16bf a = load_a_frag(Hprev, HID, mbase, k0, lane);
    #pragma unroll
    for (int g = 0; g < 4; ++g) {
      v16bf b = load_b_frag(Whh, HID, g * HID + ub, k0, lane);
      acc[g] = wmma_bf16f32(a, b, acc[g]);
    }
  }

  // ---- epilogue: i,f,g,o -> c,h (D layout: m = r + 8*(lane>=16), n = lane&15)
  const int n = lane & 15;
  const int j = ub + n;
  const int mofs = mbase + ((lane & 16) ? 8 : 0);
  const float bi = b_ih[j]           + b_hh[j];
  const float bf = b_ih[HID   + j]   + b_hh[HID   + j];
  const float bg = b_ih[2*HID + j]   + b_hh[2*HID + j];
  const float bo = b_ih[3*HID + j]   + b_hh[3*HID + j];
  #pragma unroll
  for (int r = 0; r < 8; ++r) {
    const int m = mofs + r;
    const float iv = sigmoidf_(acc[0][r] + bi);
    const float fv = sigmoidf_(acc[1][r] + bf);
    const float gv = tanhf    (acc[2][r] + bg);
    const float ov = sigmoidf_(acc[3][r] + bo);
    const size_t idx = (size_t)m * HID + j;
    const float cn = fv * Cst[idx] + iv * gv;
    Cst[idx]  = cn;
    Hout[idx] = (__bf16)(ov * tanhf(cn));
  }
}

// ---------------------------------------------------------------------------
// q = h3 @ w_proj1^T + b   (M=64, N=512, K=1024).  Grid: 8 blocks x 256 thr.
// Wave w: ntile = w>>1, mtiles {(w&1)*2, (w&1)*2+1} -> B fragment reused.
// ---------------------------------------------------------------------------
__global__ void __launch_bounds__(256)
k_proj1(const __bf16* __restrict__ H,       // BATCH x HID bf16
        const __bf16* __restrict__ Wp,      // CTX x HID bf16
        const float*  __restrict__ bp,      // CTX
        float*        __restrict__ Q)       // BATCH x CTX f32
{
  const int lane = threadIdx.x & 31;
  const int w    = threadIdx.x >> 5;
  const int colbase = blockIdx.x * 64 + (w >> 1) * 16;
  const int m0 = ((w & 1) * 2) * 16;
  const int m1 = m0 + 16;

  v8f acc0 = v8f{0.f,0.f,0.f,0.f,0.f,0.f,0.f,0.f};
  v8f acc1 = acc0;
  for (int k0 = 0; k0 < HID; k0 += 32) {
    v16bf b  = load_b_frag(Wp, HID, colbase, k0, lane);
    v16bf a0 = load_a_frag(H, HID, m0, k0, lane);
    v16bf a1 = load_a_frag(H, HID, m1, k0, lane);
    acc0 = wmma_bf16f32(a0, b, acc0);
    acc1 = wmma_bf16f32(a1, b, acc1);
  }
  const int col = colbase + (lane & 15);
  const float bias = bp[col];
  const int mo = (lane & 16) ? 8 : 0;
  #pragma unroll
  for (int r = 0; r < 8; ++r) {
    Q[(size_t)(m0 + mo + r) * CTX + col] = acc0[r] + bias;
    Q[(size_t)(m1 + mo + r) * CTX + col] = acc1[r] + bias;
  }
}

// ---------------------------------------------------------------------------
// Per-batch attention + context + vocab projection + next-step x build.
// One block (512 thr) per batch element b.
//   energy[s] = sum_c q[c] * Kflat[b, c*S + s]   (reshape semantics!)
//   softmax -> *mask -> L1 normalize -> ctx = attn @ V_b
//   out[t+1,b,:] = h3_b @ w_proj2^T + b_proj2
//   xbuf_next = [ embedding[target[t+1,b]] | ctx ]  (bf16)
// ---------------------------------------------------------------------------
__global__ void __launch_bounds__(512)
k_attn(const float*  __restrict__ Q,        // BATCH x CTX
       const __bf16* __restrict__ Kbf,      // BATCH x (S*C) flat bf16
       const __bf16* __restrict__ Vbf,      // BATCH x (S*C) flat bf16
       const float*  __restrict__ mask,     // BATCH x S
       const int*    __restrict__ target,   // T x BATCH
       const float*  __restrict__ emb,      // VOCAB x EMB
       const __bf16* __restrict__ H3,       // BATCH x HID bf16
       const __bf16* __restrict__ Wp2,      // VOCAB x HID bf16
       const float*  __restrict__ bp2,      // VOCAB
       __bf16*       __restrict__ xbuf,     // BATCH x (EMB+CTX) bf16
       float*        __restrict__ out,      // T*BATCH x VOCAB
       int t)
{
  const int b   = blockIdx.x;
  const int tid = threadIdx.x;
  __shared__ float sq[CTX];
  __shared__ float sred[512];
  __shared__ float sattn[512];

  sq[tid] = Q[(size_t)b * CTX + tid];
  __syncthreads();

  // energy (threads 0..399)
  float e = 0.f;
  if (tid < S_ENC) {
    const __bf16* Kb = Kbf + (size_t)b * (S_ENC * CTX);
    #pragma unroll 4
    for (int c = 0; c < CTX; ++c)
      e += sq[c] * (float)Kb[(size_t)c * S_ENC + tid];
  }

  // max reduce
  sred[tid] = (tid < S_ENC) ? e : -3.4e38f;
  __syncthreads();
  for (int s = 256; s > 0; s >>= 1) {
    if (tid < s) sred[tid] = fmaxf(sred[tid], sred[tid + s]);
    __syncthreads();
  }
  const float emax = sred[0];
  __syncthreads();

  float ex = (tid < S_ENC) ? __expf(e - emax) : 0.f;
  sred[tid] = ex;
  __syncthreads();
  for (int s = 256; s > 0; s >>= 1) {
    if (tid < s) sred[tid] += sred[tid + s];
    __syncthreads();
  }
  const float esum = sred[0];
  __syncthreads();

  float am = 0.f;
  if (tid < S_ENC) am = (ex / esum) * mask[(size_t)b * S_ENC + tid];
  sred[tid] = fabsf(am);
  __syncthreads();
  for (int s = 256; s > 0; s >>= 1) {
    if (tid < s) sred[tid] += sred[tid + s];
    __syncthreads();
  }
  const float denom = fmaxf(sred[0], 1e-12f);
  sattn[tid] = (tid < S_ENC) ? (am / denom) : 0.f;
  __syncthreads();

  // context: ctx[c] = sum_s attn[s] * V[b,s,c]   (all 512 threads)
  {
    const __bf16* Vb = Vbf + (size_t)b * (S_ENC * CTX);
    float acc = 0.f;
    #pragma unroll 4
    for (int s = 0; s < S_ENC; ++s)
      acc += sattn[s] * (float)Vb[(size_t)s * CTX + tid];
    xbuf[(size_t)b * KIN1 + EMB + tid] = (__bf16)acc;
  }

  // next-step embedding part of x
  if (tid < EMB) {
    const int y = target[(size_t)(t + 1) * BATCH + b];
    xbuf[(size_t)b * KIN1 + tid] = (__bf16)emb[(size_t)y * EMB + tid];
  }

  // vocab projection out[t+1, b, :]
  if (tid < VOCAB) {
    const __bf16* h = H3 + (size_t)b * HID;
    const __bf16* wr = Wp2 + (size_t)tid * HID;
    float o = bp2[tid];
    #pragma unroll 8
    for (int k = 0; k < HID; ++k) o += (float)h[k] * (float)wr[k];
    out[((size_t)(t + 1) * BATCH + b) * VOCAB + tid] = o;
  }
}

// ---------------------------------------------------------------------------
// Prepass utility kernels
// ---------------------------------------------------------------------------
__global__ void k_cvt_bf16(const float* __restrict__ src, __bf16* __restrict__ dst, int n) {
  int i = blockIdx.x * blockDim.x + threadIdx.x;
  for (; i < n; i += gridDim.x * blockDim.x) dst[i] = (__bf16)src[i];
}
__global__ void k_zero_u32(unsigned int* __restrict__ p, int nwords) {
  int i = blockIdx.x * blockDim.x + threadIdx.x;
  for (; i < nwords; i += gridDim.x * blockDim.x) p[i] = 0u;
}
__global__ void k_init_xbuf(const int* __restrict__ target, const float* __restrict__ emb,
                            __bf16* __restrict__ xbuf) {
  int i = blockIdx.x * blockDim.x + threadIdx.x;
  const int n = BATCH * KIN1;
  for (; i < n; i += gridDim.x * blockDim.x) {
    const int b = i / KIN1, col = i % KIN1;
    float v = 0.f;
    if (col < EMB) v = emb[(size_t)target[b] * EMB + col];   // target[0, b]
    xbuf[i] = (__bf16)v;
  }
}

// ---------------------------------------------------------------------------
// Host driver
// ---------------------------------------------------------------------------
extern "C" void kernel_launch(void* const* d_in, const int* in_sizes, int n_in,
                              void* d_out, int out_size, void* d_ws, size_t ws_size,
                              hipStream_t stream) {
  (void)in_sizes; (void)n_in; (void)out_size; (void)ws_size;
  const int*   target  = (const int*)  d_in[0];
  // d_in[1] target_lengths: unused by reference
  const float* attK    = (const float*)d_in[2];
  const float* attV    = (const float*)d_in[3];
  const float* attMask = (const float*)d_in[4];
  const float* embedF  = (const float*)d_in[5];
  const float* w_ih1 = (const float*)d_in[6];  const float* w_hh1 = (const float*)d_in[7];
  const float* b_ih1 = (const float*)d_in[8];  const float* b_hh1 = (const float*)d_in[9];
  const float* w_ih2 = (const float*)d_in[10]; const float* w_hh2 = (const float*)d_in[11];
  const float* b_ih2 = (const float*)d_in[12]; const float* b_hh2 = (const float*)d_in[13];
  const float* w_ih3 = (const float*)d_in[14]; const float* w_hh3 = (const float*)d_in[15];
  const float* b_ih3 = (const float*)d_in[16]; const float* b_hh3 = (const float*)d_in[17];
  const float* w_proj1 = (const float*)d_in[18]; const float* b_proj1 = (const float*)d_in[19];
  const float* w_proj2 = (const float*)d_in[20]; const float* b_proj2 = (const float*)d_in[21];
  float* out = (float*)d_out;

  // ---- workspace layout (bytes, all 256-aligned) ----
  char* ws = (char*)d_ws;
  size_t off = 0;
  auto take = [&](size_t bytes) { char* p = ws + off; off += (bytes + 255) & ~(size_t)255; return p; };
  __bf16* wih1b = (__bf16*)take((size_t)G4 * KIN1 * 2);
  __bf16* whh1b = (__bf16*)take((size_t)G4 * HID  * 2);
  __bf16* wih2b = (__bf16*)take((size_t)G4 * HID  * 2);
  __bf16* whh2b = (__bf16*)take((size_t)G4 * HID  * 2);
  __bf16* wih3b = (__bf16*)take((size_t)G4 * HID  * 2);
  __bf16* whh3b = (__bf16*)take((size_t)G4 * HID  * 2);
  __bf16* wp1b  = (__bf16*)take((size_t)CTX * HID * 2);
  __bf16* wp2b  = (__bf16*)take((size_t)VOCAB * HID * 2);
  __bf16* Kbf   = (__bf16*)take((size_t)KV_ELEMS * 2);
  __bf16* Vbf   = (__bf16*)take((size_t)KV_ELEMS * 2);
  __bf16* hbf   = (__bf16*)take((size_t)6 * BATCH * HID * 2);  // h1/h2/h3 x 2 parities
  float*  cst   = (float*) take((size_t)3 * BATCH * HID * 4);  // c1,c2,c3
  __bf16* xbuf  = (__bf16*)take((size_t)BATCH * KIN1 * 2);
  float*  q     = (float*) take((size_t)BATCH * CTX * 4);

  __bf16* h1 = hbf;                       // [2][BATCH*HID]
  __bf16* h2 = hbf + 2 * BATCH * HID;
  __bf16* h3 = hbf + 4 * BATCH * HID;
  float *c1 = cst, *c2 = cst + BATCH * HID, *c3 = cst + 2 * BATCH * HID;

  // ---- prepass: bf16 conversions ----
  auto cvt = [&](const float* s, __bf16* d, int n) {
    int blk = (n + 255) / 256; if (blk > 16384) blk = 16384;
    k_cvt_bf16<<<blk, 256, 0, stream>>>(s, d, n);
  };
  cvt(w_ih1, wih1b, G4 * KIN1);
  cvt(w_hh1, whh1b, G4 * HID);
  cvt(w_ih2, wih2b, G4 * HID);
  cvt(w_hh2, whh2b, G4 * HID);
  cvt(w_ih3, wih3b, G4 * HID);
  cvt(w_hh3, whh3b, G4 * HID);
  cvt(w_proj1, wp1b, CTX * HID);
  cvt(w_proj2, wp2b, VOCAB * HID);
  cvt(attK, Kbf, KV_ELEMS);
  cvt(attV, Vbf, KV_ELEMS);

  // zero h (both parities), c, and out row 0
  k_zero_u32<<<((6 * BATCH * HID * 2) / 4 + 255) / 256, 256, 0, stream>>>(
      (unsigned int*)hbf, (6 * BATCH * HID * 2) / 4);
  k_zero_u32<<<((3 * BATCH * HID * 4) / 4 + 255) / 256, 256, 0, stream>>>(
      (unsigned int*)cst, 3 * BATCH * HID);
  k_zero_u32<<<(BATCH * VOCAB + 255) / 256, 256, 0, stream>>>(
      (unsigned int*)out, BATCH * VOCAB);
  k_init_xbuf<<<(BATCH * KIN1 + 255) / 256, 256, 0, stream>>>(target, embedF, xbuf);

  // ---- 299 decoder steps ----
  const int HB = BATCH * HID;
  for (int t = 0; t < T_STEPS - 1; ++t) {
    const int p = t & 1, np = p ^ 1;
    __bf16 *h1p = h1 + p * HB, *h1n = h1 + np * HB;
    __bf16 *h2p = h2 + p * HB, *h2n = h2 + np * HB;
    __bf16 *h3p = h3 + p * HB, *h3n = h3 + np * HB;

    k_lstm<<<HID / 32, 256, 0, stream>>>(xbuf, KIN1, wih1b, h1p, whh1b,
                                         b_ih1, b_hh1, c1, h1n);
    k_lstm<<<HID / 32, 256, 0, stream>>>(h1n, HID, wih2b, h2p, whh2b,
                                         b_ih2, b_hh2, c2, h2n);
    k_lstm<<<HID / 32, 256, 0, stream>>>(h2n, HID, wih3b, h3p, whh3b,
                                         b_ih3, b_hh3, c3, h3n);
    k_proj1<<<CTX / 64, 256, 0, stream>>>(h3n, wp1b, b_proj1, q);
    k_attn<<<BATCH, 512, 0, stream>>>(q, Kbf, Vbf, attMask, target, embedF,
                                      h3n, wp2b, b_proj2, xbuf, out, t);
  }
}
